// TemporalMambaBlock_85856396247380
// MI455X (gfx1250) — matrix operand order repaired and checked by reading, not compiled, not run on hardware
//
#include <hip/hip_runtime.h>
#include <hip/hip_bf16.h>

typedef __attribute__((ext_vector_type(16))) __bf16 v16bf;
typedef __attribute__((ext_vector_type(8)))  float  v8f;
typedef __attribute__((ext_vector_type(4)))  unsigned v4u;
typedef __attribute__((ext_vector_type(8)))  int    v8i;
typedef __attribute__((ext_vector_type(4)))  int    v4i;

#define D_MODEL 256
#define D_STATE 16
#define D_CONV  4
#define D_INNER 512
#define DT_RANK 16
#define SEQ_B 2
#define SEQ_T 16
#define SEQ_H 24
#define SEQ_W 24
#define NSEQ (SEQ_B*SEQ_H*SEQ_W)      /* 1152 */
#define MTOT (NSEQ*SEQ_T)             /* 18432 rows of (n,t) */

/* -------- small helpers -------- */
static __device__ __forceinline__ unsigned short f2bf(float f) {
  unsigned u = __float_as_uint(f);
  u += 0x7FFFu + ((u >> 16) & 1u);           /* round-to-nearest-even */
  return (unsigned short)(u >> 16);
}
static __device__ __forceinline__ float bf2f(unsigned short h) {
  return __uint_as_float(((unsigned)h) << 16);
}
static __device__ __forceinline__ float silu(float x) {
  return x / (1.f + __expf(-x));
}
static __device__ __forceinline__ float softplus(float x) {
  return (x > 20.f) ? x : __logf(1.f + __expf(x));
}

/* A-fragment (16x32 bf16, row-major source A[m][k], stride in dwords). */
static __device__ __forceinline__ v16bf
load_a_frag(const unsigned* base_dw, int stride_dw, int row, int kbase, int lane) {
  int r  = row + (lane & 15);
  int kh = (lane < 16) ? 0 : 8;
  const unsigned* p = base_dw + (size_t)r * stride_dw;
  union { v16bf v; unsigned u[8]; } f;
#pragma unroll
  for (int i = 0; i < 8; ++i) {
    int k = kbase + ((i < 4) ? (kh + 2*i) : (16 + kh + 2*(i - 4)));
    f.u[i] = p[k >> 1];
  }
  return f.v;
}

/* B-fragment (32x16 bf16), source W[n][k] row-major (computing X*W^T). */
static __device__ __forceinline__ v16bf
load_b_frag(const unsigned* base_dw, int stride_dw, int row, int kbase, int lane) {
  int r  = row + (lane & 15);
  int kh = (lane < 16) ? 0 : 8;
  const unsigned* p = base_dw + (size_t)r * stride_dw;
  union { v16bf v; unsigned u[8]; } f;
#pragma unroll
  for (int i = 0; i < 8; ++i) {
    int k = kbase + ((i < 4) ? (kh + 2*i) : (16 + kh + 2*(i - 4)));
    f.u[i] = p[k >> 1];
  }
  return f.v;
}

static __device__ __forceinline__ v8f wmma_bf16(v16bf a, v16bf b, v8f c) {
  return __builtin_amdgcn_wmma_f32_16x16x32_bf16(false, a, false, b, (short)0, c, false, false);
}

/* -------- kernel 0: gather/transpose x_seq -> Xb bf16 (MTOT x 256) and
   compact last-timestep matrix Xl bf16 (NSEQ x 256) -------- */
__global__ __launch_bounds__(256) void k_pack(const float* __restrict__ x_seq,
                                              unsigned short* __restrict__ Xb,
                                              unsigned short* __restrict__ Xl) {
  int m = blockIdx.x;          /* 0..18431 */
  int c = threadIdx.x;         /* 0..255  */
  int t = m & 15;
  int n = m >> 4;
  int w = n % SEQ_W;
  int h = (n / SEQ_W) % SEQ_H;
  int b = n / (SEQ_H * SEQ_W);
  size_t src = ((((size_t)b * SEQ_T + t) * D_MODEL + c) * SEQ_H + h) * SEQ_W + w;
  unsigned short v = f2bf(x_seq[src]);
  Xb[(size_t)m * D_MODEL + c] = v;
  if (t == SEQ_T - 1) Xl[(size_t)n * D_MODEL + c] = v;
}

/* -------- kernel 1: f32 -> bf16 weight conversion -------- */
__global__ __launch_bounds__(256) void k_f2bf(const float* __restrict__ src,
                                              unsigned short* __restrict__ dst, int n) {
  int i = blockIdx.x * 256 + threadIdx.x;
  if (i < n) dst[i] = f2bf(src[i]);
}

/* -------- TDM staging: contiguous (128 x 256 bf16) chunk -> padded-stride LDS.
   2D tile: tile_dim0=256 elems (512B = 128 dwords/row), tile_dim1=128 rows,
   pad 4 dwords after every 128 dwords -> LDS row stride 132 dwords. -------- */
static __device__ __forceinline__ void
tdm_load_w_chunk(const unsigned short* gsrc, void* lds_dst) {
  unsigned long long ga = (unsigned long long)(size_t)gsrc;
  unsigned lds_off = (unsigned)(size_t)lds_dst;   /* LDS aperture: addr[31:0] = offset */
  v4u g0;
  g0[0] = 1u;                                       /* count=1, user descriptor */
  g0[1] = lds_off;                                  /* lds_addr */
  g0[2] = (unsigned)(ga & 0xFFFFFFFFu);             /* global_addr[31:0] */
  g0[3] = (unsigned)((ga >> 32) & 0x1FFFFFFu)       /* global_addr[56:32] */
        | (2u << 30);                               /* type = 2 (image) */
  v8i g1;
  g1[0] = (int)((1u << 16)      /* data_size = 2 bytes */
              | (1u << 20)      /* pad_enable */
              | (6u << 22)      /* pad_interval: 128 dwords */
              | (3u << 25));    /* pad_amount:   4 dwords  */
  g1[1] = (int)(256u << 16);    /* tensor_dim0[15:0]=256 (atomic_barrier_addr=0) */
  g1[2] = (int)(128u << 16);    /* tensor_dim0[31:16]=0 | tensor_dim1[15:0]=128 */
  g1[3] = (int)(256u << 16);    /* tensor_dim1[31:16]=0 | tile_dim0=256 */
  g1[4] = (int)(128u);          /* tile_dim1=128 | tile_dim2=0 */
  g1[5] = (int)(256);           /* tensor_dim0_stride[31:0]=256 */
  g1[6] = 0;                    /* stride0 hi | stride1 lo */
  g1[7] = 0;                    /* stride1 hi */
  v4i g2 = {0, 0, 0, 0};
  v4i g3 = {0, 0, 0, 0};
#if defined(__clang_major__) && (__clang_major__ >= 23)
  v8i g1b = {0, 0, 0, 0, 0, 0, 0, 0};
  __builtin_amdgcn_tensor_load_to_lds(g0, g1, g2, g3, g1b, 0);
#else
  __builtin_amdgcn_tensor_load_to_lds(g0, g1, g2, g3, 0);
#endif
  __builtin_amdgcn_s_wait_tensorcnt((short)0);
}

/* -------- kernel 2: in_proj (xc half only) GEMM + causal conv(4) + SiLU --------
   Per workgroup: 8 sequences -> 128 contiguous rows of Xb.  256 threads = 8 waves.
   4 chunks of 128 output channels (W rows 0..511); wave w owns 16 channels.
   Conv along t is done fully in registers: only t=8,9,10 need values from the
   other lane-half (t=5,6,7), fetched with three ds_bpermute shuffles. */
#define XSTR 132  /* padded LDS row stride in dwords (256 bf16 + 4 pad dwords) */
__global__ __launch_bounds__(256) void k_inproj(const unsigned short* __restrict__ Xb,
                                                const unsigned short* __restrict__ W1b,
                                                const float* __restrict__ conv_w,
                                                const float* __restrict__ conv_b,
                                                unsigned short* __restrict__ xc) {
  extern __shared__ char smem[];
  unsigned* sX = (unsigned*)smem;                       /* 128*132 dw = 67584 B */
  unsigned* sW = sX + 128 * XSTR;                       /* 67584 B             */

  int tid  = threadIdx.x;
  int wave = tid >> 5;
  int lane = tid & 31;
  int n0   = blockIdx.x * 8;
  int row0 = n0 * 16;

  /* stage X tile (contiguous rows of Xb) into padded LDS via plain vmem */
  const unsigned* gX = (const unsigned*)Xb + (size_t)row0 * 128;
  for (int idx = tid; idx < 128 * 128; idx += 256) {
    int r = idx >> 7, c = idx & 127;
    sX[r * XSTR + c] = gX[idx];
  }

  for (int nc = 0; nc < 4; ++nc) {
    __syncthreads();
    const unsigned short* gW = W1b + (size_t)(nc * 128) * 256;
    if (wave == 0) tdm_load_w_chunk(gW, sW);            /* TDM DMA -> LDS */
    if (nc < 3) {                                       /* prefetch next chunk to L2 */
      const char* gWn = (const char*)(gW + 128 * 256);
      for (int p = tid; p < 512; p += 256) __builtin_prefetch(gWn + p * 128, 0, 1);
    }
    __syncthreads();

    int cglob = nc * 128 + wave * 16;        /* wave's output-channel base (<512) */
    int c_l   = lane & 15;
    int c     = cglob + c_l;
    float cw0 = conv_w[c * 4 + 0], cw1 = conv_w[c * 4 + 1];
    float cw2 = conv_w[c * 4 + 2], cw3 = conv_w[c * 4 + 3];
    float cb  = conv_b[c];
    bool  hi  = (lane >= 16);
    int   toff = hi ? 8 : 0;

    /* B fragments for this wave's 16 columns, all 8 K-steps (reused over 8 M tiles) */
    v16bf bfrag[8];
#pragma unroll
    for (int ks = 0; ks < 8; ++ks)
      bfrag[ks] = load_b_frag(sW, XSTR, wave * 16, ks * 32, lane);

    for (int mt = 0; mt < 8; ++mt) {         /* one M tile = one sequence's 16 t */
      v16bf afrag[8];
#pragma unroll
      for (int ks = 0; ks < 8; ++ks)
        afrag[ks] = load_a_frag(sX, XSTR, mt * 16, ks * 32, lane);
      v8f acc = {0.f,0.f,0.f,0.f,0.f,0.f,0.f,0.f};
#pragma unroll
      for (int ks = 0; ks < 8; ++ks)
        acc = wmma_bf16(afrag[ks], bfrag[ks], acc);

      /* cross-half history: lanes>=16 computing t=8,9,10 need t=5,6,7 from lane-16 */
      float s5 = __shfl(acc[5], c_l, 32);
      float s6 = __shfl(acc[6], c_l, 32);
      float s7 = __shfl(acc[7], c_l, 32);
      float e[11];
      e[0] = hi ? s5 : 0.f;   /* value at local t-3 of r=0 */
      e[1] = hi ? s6 : 0.f;
      e[2] = hi ? s7 : 0.f;
#pragma unroll
      for (int r = 0; r < 8; ++r) e[r + 3] = acc[r];

      int n = n0 + mt;
#pragma unroll
      for (int r = 0; r < 8; ++r) {
        int t = r + toff;
        float s = cb + cw0 * e[r] + cw1 * e[r + 1] + cw2 * e[r + 2] + cw3 * e[r + 3];
        xc[((size_t)(n * 16 + t)) * D_INNER + c] = f2bf(silu(s));
      }
    }
  }
}

/* -------- kernel 2b: z projection, last timestep only --------
   Xl(1152x256) * W1b[512:1024]^T -> z_last f32 (1152x512). */
__global__ __launch_bounds__(256) void k_zproj(const unsigned short* __restrict__ Xl,
                                               const unsigned short* __restrict__ W1b,
                                               float* __restrict__ z_last) {
  int wave = threadIdx.x >> 5;               /* 0..7 */
  int lane = threadIdx.x & 31;
  int mt   = blockIdx.x;                     /* 0..71 */
  int nt   = blockIdx.y * 8 + wave;          /* 0..31 */
  v8f acc = {0.f,0.f,0.f,0.f,0.f,0.f,0.f,0.f};
#pragma unroll
  for (int ks = 0; ks < 8; ++ks) {
    v16bf a = load_a_frag((const unsigned*)Xl, D_MODEL / 2, mt * 16, ks * 32, lane);
    v16bf b = load_b_frag((const unsigned*)W1b, D_MODEL / 2, D_INNER + nt * 16, ks * 32, lane);
    acc = wmma_bf16(a, b, acc);
  }
#pragma unroll
  for (int r = 0; r < 8; ++r) {
    int n = mt * 16 + r + ((lane < 16) ? 0 : 8);
    int c = nt * 16 + (lane & 15);
    z_last[(size_t)n * D_INNER + c] = acc[r];
  }
}

/* -------- kernel 3: x_proj GEMM  xc(18432x512) * Wx^T(512x48) -> x_dbl f32 ----
   Wave preloads all 16 B K-step fragments once and sweeps 4 M tiles. -------- */
__global__ __launch_bounds__(96) void k_xproj(const unsigned short* __restrict__ xc,
                                              const unsigned short* __restrict__ Wxb,
                                              float* __restrict__ x_dbl) {
  int wave = threadIdx.x >> 5;   /* 0..2 -> n-tile */
  int lane = threadIdx.x & 31;
  int mt0  = blockIdx.x * 4;     /* 4 M tiles per block */
  v16bf bfrag[16];
#pragma unroll
  for (int ks = 0; ks < 16; ++ks)
    bfrag[ks] = load_b_frag((const unsigned*)Wxb, D_INNER / 2, wave * 16, ks * 32, lane);
  for (int i = 0; i < 4; ++i) {
    int mt = mt0 + i;
    v8f acc = {0.f,0.f,0.f,0.f,0.f,0.f,0.f,0.f};
#pragma unroll
    for (int ks = 0; ks < 16; ++ks) {
      v16bf a = load_a_frag((const unsigned*)xc, D_INNER / 2, mt * 16, ks * 32, lane);
      acc = wmma_bf16(a, bfrag[ks], acc);
    }
#pragma unroll
    for (int r = 0; r < 8; ++r) {
      int m = mt * 16 + r + ((lane < 16) ? 0 : 8);
      int c = wave * 16 + (lane & 15);
      x_dbl[(size_t)m * 48 + c] = acc[r];
    }
  }
}

/* -------- kernel 4: dt_proj + softplus + SSM scan (h in registers) -------- */
__global__ __launch_bounds__(256) void k_scan(const unsigned short* __restrict__ xc,
                                              const float* __restrict__ x_dbl,
                                              const float* __restrict__ z_last,
                                              const float* __restrict__ dt_w,
                                              const float* __restrict__ dt_b,
                                              const float* __restrict__ A_log,
                                              const float* __restrict__ Dv,
                                              unsigned short* __restrict__ y_last) {
  __shared__ float srow[48];
  int n   = blockIdx.x;
  int tid = threadIdx.x;
  int d0  = tid * 2;

  float h[2][D_STATE], Am[2][D_STATE], dw[2][D_STATE], db[2], Dd[2];
#pragma unroll
  for (int c = 0; c < 2; ++c) {
    int d = d0 + c;
    db[c] = dt_b[d];
    Dd[c] = Dv[d];
#pragma unroll
    for (int s = 0; s < D_STATE; ++s) {
      Am[c][s] = -__expf(A_log[d * D_STATE + s]);
      dw[c][s] = dt_w[d * DT_RANK + s];
      h[c][s]  = 0.f;
    }
  }

  for (int t = 0; t < SEQ_T; ++t) {
    __syncthreads();
    if (tid < 48) srow[tid] = x_dbl[(size_t)(n * 16 + t) * 48 + tid];
    __syncthreads();
    unsigned xp = *(const unsigned*)(xc + (size_t)(n * 16 + t) * D_INNER + d0);
#pragma unroll
    for (int c = 0; c < 2; ++c) {
      int d = d0 + c;
      float xv = bf2f((unsigned short)((c == 0) ? (xp & 0xFFFFu) : (xp >> 16)));
      float a  = db[c];
#pragma unroll
      for (int r = 0; r < DT_RANK; ++r) a += srow[r] * dw[c][r];
      float dt = softplus(a);
      float dx = dt * xv;
      if (t == SEQ_T - 1) {
        float ys = 0.f;
#pragma unroll
        for (int s = 0; s < D_STATE; ++s) {
          float dA = __expf(dt * Am[c][s]);
          h[c][s] = dA * h[c][s] + dx * srow[16 + s];
          ys += h[c][s] * srow[32 + s];
        }
        float z  = z_last[(size_t)n * D_INNER + d];
        float yv = (ys + xv * Dd[c]) * silu(z);
        y_last[(size_t)n * D_INNER + d] = f2bf(yv);
      } else {
#pragma unroll
        for (int s = 0; s < D_STATE; ++s) {
          float dA = __expf(dt * Am[c][s]);
          h[c][s] = dA * h[c][s] + dx * srow[16 + s];
        }
      }
    }
  }
}

/* -------- kernel 5: out_proj GEMM y_last(1152x512) * Wo^T(512x256), scatter NCHW ---- */
__global__ __launch_bounds__(512) void k_outproj(const unsigned short* __restrict__ y_last,
                                                 const unsigned short* __restrict__ Wob,
                                                 float* __restrict__ out) {
  int wave = threadIdx.x >> 5;   /* 0..15 -> n-tile */
  int lane = threadIdx.x & 31;
  int mt   = blockIdx.x;         /* 0..71 */
  v8f acc = {0.f,0.f,0.f,0.f,0.f,0.f,0.f,0.f};
#pragma unroll
  for (int ks = 0; ks < 16; ++ks) {
    v16bf a = load_a_frag((const unsigned*)y_last, D_INNER / 2, mt * 16, ks * 32, lane);
    v16bf b = load_b_frag((const unsigned*)Wob, D_INNER / 2, wave * 16, ks * 32, lane);
    acc = wmma_bf16(a, b, acc);
  }
#pragma unroll
  for (int r = 0; r < 8; ++r) {
    int m   = mt * 16 + r + ((lane < 16) ? 0 : 8);   /* sequence index n */
    int c   = wave * 16 + (lane & 15);               /* output channel   */
    int b   = m / (SEQ_H * SEQ_W);
    int rem = m % (SEQ_H * SEQ_W);
    int hh  = rem / SEQ_W;
    int ww  = rem % SEQ_W;
    out[(((size_t)b * D_MODEL + c) * SEQ_H + hh) * SEQ_W + ww] = acc[r];
  }
}

/* -------- workspace layout (bytes) -------- */
#define OFF_XB    ((size_t)0)                          /* 18432*256*2 = 9437184  */
#define OFF_W1B   ((size_t)9437184)                    /* 1024*256*2  = 524288   */
#define OFF_WXB   ((size_t)9961472)                    /* 48*512*2    = 49152    */
#define OFF_WOB   ((size_t)10010624)                   /* 256*512*2   = 262144   */
#define OFF_XC    ((size_t)10272768)                   /* 18432*512*2 = 18874368 */
#define OFF_ZL    ((size_t)29147136)                   /* 1152*512*4  = 2359296  */
#define OFF_XDBL  ((size_t)31506432)                   /* 18432*48*4  = 3538944  */
#define OFF_YL    ((size_t)35045376)                   /* 1152*512*2  = 1179648  */
#define OFF_XL    ((size_t)36225024)                   /* 1152*256*2  = 589824   */

extern "C" void kernel_launch(void* const* d_in, const int* in_sizes, int n_in,
                              void* d_out, int out_size, void* d_ws, size_t ws_size,
                              hipStream_t stream) {
  const float* x_seq     = (const float*)d_in[0];
  const float* in_proj_w = (const float*)d_in[1];
  const float* conv_w    = (const float*)d_in[2];
  const float* conv_b    = (const float*)d_in[3];
  const float* x_proj_w  = (const float*)d_in[4];
  const float* dt_proj_w = (const float*)d_in[5];
  const float* dt_proj_b = (const float*)d_in[6];
  const float* A_log     = (const float*)d_in[7];
  const float* Dvec      = (const float*)d_in[8];
  const float* out_projw = (const float*)d_in[9];
  float* out = (float*)d_out;
  char*  ws  = (char*)d_ws;

  unsigned short* Xb  = (unsigned short*)(ws + OFF_XB);
  unsigned short* W1b = (unsigned short*)(ws + OFF_W1B);
  unsigned short* Wxb = (unsigned short*)(ws + OFF_WXB);
  unsigned short* Wob = (unsigned short*)(ws + OFF_WOB);
  unsigned short* xc  = (unsigned short*)(ws + OFF_XC);
  float*          zl  = (float*)(ws + OFF_ZL);
  float*          xd  = (float*)(ws + OFF_XDBL);
  unsigned short* yl  = (unsigned short*)(ws + OFF_YL);
  unsigned short* Xl  = (unsigned short*)(ws + OFF_XL);

  k_pack<<<MTOT, 256, 0, stream>>>(x_seq, Xb, Xl);
  k_f2bf<<<(1024 * 256 + 255) / 256, 256, 0, stream>>>(in_proj_w, W1b, 1024 * 256);
  k_f2bf<<<(48 * 512 + 255) / 256, 256, 0, stream>>>(x_proj_w, Wxb, 48 * 512);
  k_f2bf<<<(256 * 512 + 255) / 256, 256, 0, stream>>>(out_projw, Wob, 256 * 512);

  size_t lds_bytes = (size_t)(2 * 128 * XSTR * 4);     /* 135168 */
  k_inproj<<<NSEQ / 8, 256, lds_bytes, stream>>>(Xb, W1b, conv_w, conv_b, xc);
  k_zproj<<<dim3(NSEQ / 16, 4), 256, 0, stream>>>(Xl, W1b, zl);

  k_xproj<<<MTOT / 16 / 4, 96, 0, stream>>>(xc, Wxb, xd);
  k_scan<<<NSEQ, 256, 0, stream>>>(xc, xd, zl, dt_proj_w, dt_proj_b, A_log, Dvec, yl);
  k_outproj<<<NSEQ / 16, 512, 0, stream>>>(yl, Wob, out);
}
